// MlutiLatentAtten_678604833165
// MI455X (gfx1250) — matrix operand order repaired and checked by reading, not compiled, not run on hardware
//
#include <hip/hip_runtime.h>

// ---------------------------------------------------------------------------
// MI455X (gfx1250) multi-head attention, bf16 WMMA pipeline.
//  - all GEMMs through v_wmma_f32_16x16x32_bf16
//  - attention K/V tiles staged by the Tensor Data Mover (tensor_load_to_lds)
//  - GEMM tiles staged with async global->LDS copies
// ---------------------------------------------------------------------------

typedef __bf16 bf16;
typedef __attribute__((ext_vector_type(16))) __bf16 v16bf;
typedef __attribute__((ext_vector_type(8)))  float  v8f;
typedef __attribute__((ext_vector_type(4)))  unsigned int u32x4;
typedef __attribute__((ext_vector_type(8)))  int i32x8;
typedef __attribute__((ext_vector_type(4)))  int i32x4;

union Frag16 {
  v16bf    v;
  bf16     e[16];
  uint4    q[2];
  unsigned u[8];
};

#define WMMA_BF16(a, b, c) \
  __builtin_amdgcn_wmma_f32_16x16x32_bf16(false, (a), false, (b), (short)0, (c), false, false)

#define S_LEN   2048
#define HID     2048
#define HEADS   16
#define HD      128
#define BATCH   4
#define MROWS   (BATCH * S_LEN)      // 8192
#define LOG1E4_OVER_D (9.210340372f / 128.0f)

// ---- CDNA5 data-mover feature detection (per compilation pass) ------------
#if defined(__AMDGCN__) && __has_builtin(__builtin_amdgcn_tensor_load_to_lds) && \
    __has_builtin(__builtin_amdgcn_s_wait_tensorcnt)
#define HAVE_TDM 1
#else
#define HAVE_TDM 0
#endif

#if defined(__AMDGCN__) && __has_builtin(__builtin_amdgcn_global_load_async_to_lds_b128) && \
    __has_builtin(__builtin_amdgcn_s_wait_asynccnt)
#define HAVE_ASYNC 1
#else
#define HAVE_ASYNC 0
#endif

#if HAVE_ASYNC
// Builtin signature (from hipcc diagnostic): first param is
// 'int __vector(4) addrspace(1)*'; LDS side symmetric in addrspace(3).
typedef int v4i_t __attribute__((vector_size(16)));
typedef v4i_t __attribute__((address_space(1)))* gv4p;
typedef v4i_t __attribute__((address_space(3)))* lv4p;
// 16-byte global -> LDS async copy (tracked by ASYNCcnt).
static __device__ __forceinline__ void async_cp16(const void* g, const void* lds) {
  __builtin_amdgcn_global_load_async_to_lds_b128(
      (gv4p)(uintptr_t)g,
      (lv4p)(unsigned)(uintptr_t)lds,   // LDS aperture: addr[31:0] == LDS offset
      0, 0);
}
#endif

#if HAVE_TDM
// TDM 2D tile load: tile_dim0=128 elems (bf16), tile_dim1=32 rows, row stride
// 128 elems, contiguous into LDS.  D# bitfields per CDNA5 ISA 8.3/8.4.
static __device__ __forceinline__ void tdm_load_k32xd128(const void* gaddr,
                                                         unsigned lds_byte_off) {
  unsigned long long ga = (unsigned long long)(uintptr_t)gaddr;
  u32x4 g0;
  g0[0] = 1u;                                   // count=1, user descriptor
  g0[1] = lds_byte_off;                         // lds_addr [63:32]
  g0[2] = (unsigned)ga;                         // global_addr lo
  g0[3] = (unsigned)((ga >> 32) & 0x01FFFFFFu)  // global_addr [56:32]
          | 0x80000000u;                        // type=2 ("image") [127:126]
  i32x8 g1;
  g1[0] = (int)(1u << 16);        // workgroup_mask=0, data_size=1 (2 bytes)
  g1[1] = (int)(128u << 16);      // tensor_dim0 = 128 (lo16 in [63:48])
  g1[2] = (int)(2048u << 16);     // tensor_dim0 hi=0; tensor_dim1 = 2048
  g1[3] = (int)(128u << 16);      // tensor_dim1 hi=0; tile_dim0 = 128
  g1[4] = 32;                     // tile_dim1 = 32; tile_dim2 = 0 (2D tile)
  g1[5] = 128;                    // tensor_dim0_stride = 128 (lo32)
  g1[6] = 0;                      // stride hi; tensor_dim1_stride lo (unused)
  g1[7] = 0;
  i32x4 z4 = {0, 0, 0, 0};
#if __clang_major__ >= 23
  i32x8 z8 = {0, 0, 0, 0, 0, 0, 0, 0};
  __builtin_amdgcn_tensor_load_to_lds(g0, g1, z4, z4, z8, 0);
#else
  __builtin_amdgcn_tensor_load_to_lds(g0, g1, z4, z4, 0);
#endif
}
#endif

static __device__ __forceinline__ float half16_max(float x) {
#pragma unroll
  for (int m = 1; m <= 8; m <<= 1) x = fmaxf(x, __shfl_xor(x, m, 32));
  return x;
}
static __device__ __forceinline__ float half16_sum(float x) {
#pragma unroll
  for (int m = 1; m <= 8; m <<= 1) x += __shfl_xor(x, m, 32);
  return x;
}

// ---------------------------------------------------------------------------
// fp32 -> bf16 cast
// ---------------------------------------------------------------------------
__global__ void cast_f32_bf16(const float* __restrict__ in,
                              bf16* __restrict__ out, size_t n) {
  size_t i = (size_t)blockIdx.x * blockDim.x + threadIdx.x;
  if (i < n) out[i] = (bf16)in[i];
}

// ---------------------------------------------------------------------------
// GEMM: Y[M,N] = A[M,K] @ W[K,N] + bias.
//   mode 0: write fp32 row-major to outf.
//   mode 1: write bf16 to outq with head split: [(b*16+h)][s][d].
//   mode 2: like 1, plus interleaved RoPE on the 128-wide head dim.
// WG tile 64(M) x 256(N), K-step 32; 8 waves, each wave = 2x4 16x16 tiles.
// ---------------------------------------------------------------------------
__global__ __launch_bounds__(256) void proj_gemm(
    const bf16* __restrict__ A, const bf16* __restrict__ W,
    const float* __restrict__ bias,
    bf16* __restrict__ outq, float* __restrict__ outf, int mode) {
  __shared__ __align__(16) bf16 As[64 * 32];
  __shared__ __align__(16) bf16 Bs[32 * 256];

  const int tid   = threadIdx.x;
  const int lane  = tid & 31;
  const int wid   = tid >> 5;
  const int waveM = wid >> 2;   // 0..1
  const int waveN = wid & 3;    // 0..3
  const int m0 = blockIdx.x * 64;
  const int n0 = blockIdx.y * 256;

  v8f acc[2][4];
#pragma unroll
  for (int tm = 0; tm < 2; tm++)
#pragma unroll
    for (int tn = 0; tn < 4; tn++) {
      v8f z = {0.f, 0.f, 0.f, 0.f, 0.f, 0.f, 0.f, 0.f};
      acc[tm][tn] = z;
    }

  const int rr   = lane & 15;
  const int kb   = (lane >> 4) * 8;
  const int nn   = lane & 15;
  const int kb16 = (lane >> 4) * 16;

  for (int k0 = 0; k0 < HID; k0 += 32) {
    __syncthreads();
#if HAVE_ASYNC
    {  // A tile 64x32, one async b128 per thread
      int r = tid >> 2, cc = (tid & 3) * 8;
      async_cp16(&A[(size_t)(m0 + r) * HID + k0 + cc], &As[r * 32 + cc]);
    }
    {  // B tile 32x256, four async b128 per thread
      int r = tid >> 3, cc = (tid & 7) * 32;
      const bf16* gp = &W[(size_t)(k0 + r) * HID + n0 + cc];
      bf16* lp = &Bs[r * 256 + cc];
#pragma unroll
      for (int j = 0; j < 4; j++) async_cp16(gp + j * 8, lp + j * 8);
    }
    __builtin_amdgcn_s_wait_asynccnt(0);
#else
    {  // A tile 64x32
      int r = tid >> 2, cc = (tid & 3) * 8;
      *(uint4*)&As[r * 32 + cc] =
          *(const uint4*)&A[(size_t)(m0 + r) * HID + k0 + cc];
    }
    {  // B tile 32x256
      int r = tid >> 3, cc = (tid & 7) * 32;
      const uint4* src = (const uint4*)&W[(size_t)(k0 + r) * HID + n0 + cc];
      uint4* dst = (uint4*)&Bs[r * 256 + cc];
      dst[0] = src[0]; dst[1] = src[1]; dst[2] = src[2]; dst[3] = src[3];
    }
#endif
    if (k0 + 32 < HID) {  // hint next A tile -> global_prefetch_b8
      __builtin_prefetch(&A[(size_t)(m0 + (tid >> 2)) * HID + k0 + 32], 0, 1);
    }
    __syncthreads();

    Frag16 a[2];
#pragma unroll
    for (int tm = 0; tm < 2; tm++) {
      const bf16* p = &As[(waveM * 32 + tm * 16 + rr) * 32 + kb16];
      a[tm].q[0] = *(const uint4*)p;
      a[tm].q[1] = *(const uint4*)(p + 8);
    }
#pragma unroll
    for (int tn = 0; tn < 4; tn++) {
      Frag16 b;
      const bf16* p = &Bs[kb16 * 256 + waveN * 64 + tn * 16 + nn];
#pragma unroll
      for (int e = 0; e < 16; e++) b.e[e] = p[e * 256];
#pragma unroll
      for (int tm = 0; tm < 2; tm++)
        acc[tm][tn] = WMMA_BF16(a[tm].v, b.v, acc[tm][tn]);
    }
  }

  // epilogue
#pragma unroll
  for (int tm = 0; tm < 2; tm++)
#pragma unroll
    for (int tn = 0; tn < 4; tn++) {
      const int mbase = m0 + waveM * 32 + tm * 16 + (lane >> 4) * 8;
      const int ncol  = n0 + waveN * 64 + tn * 16 + (lane & 15);
      const float bv  = bias[ncol];
      const int d     = ncol & (HD - 1);
      const float invf = __expf(-(float)(d & ~1) * LOG1E4_OVER_D);
#pragma unroll
      for (int v = 0; v < 8; v++) {
        const int mrow = mbase + v;
        float val = acc[tm][tn][v] + bv;
        if (mode == 0) {
          outf[(size_t)mrow * HID + ncol] = val;
        } else {
          if (mode == 2) {
            float other = __shfl_xor(val, 1, 32);   // lane pair == column pair
            int s = mrow & (S_LEN - 1);
            float ang = (float)s * invf;
            float c = __cosf(ang), sn = __sinf(ang);
            val = (d & 1) ? (val * c + other * sn) : (val * c - other * sn);
          }
          int b = mrow >> 11, s = mrow & (S_LEN - 1);
          int h = ncol >> 7;
          outq[(((size_t)(b * HEADS + h) * S_LEN) + s) * HD + d] = (bf16)val;
        }
      }
    }
}

// ---------------------------------------------------------------------------
// Flash attention: one wave = 16 query rows, WG = 128 rows of one (b,h).
// K/V tiles staged by the Tensor Data Mover when available.
// ---------------------------------------------------------------------------
__global__ __launch_bounds__(256) void attn_kernel(
    const bf16* __restrict__ Q, const bf16* __restrict__ Kk,
    const bf16* __restrict__ V, const float* __restrict__ mask,
    bf16* __restrict__ O) {
  __shared__ __align__(16) bf16 Ks[32 * HD];
  __shared__ __align__(16) bf16 Vs[32 * HD];
  __shared__ __align__(16) bf16 Ps[8][16 * 32];

  const int tid  = threadIdx.x;
  const int lane = tid & 31;
  const int wid  = tid >> 5;
  const int bh   = blockIdx.x >> 4;
  const int q0   = (blockIdx.x & 15) * 128 + wid * 16;
  const float scale = 0.0220970869120796f;  // 1/sqrt(HIDDEN=2048)

  const int rr   = lane & 15;
  const int kb   = (lane >> 4) * 8;
  const int nn   = lane & 15;
  const int kb16 = (lane >> 4) * 16;
  const int rr8  = (lane >> 4) * 8;

  Frag16 qf[4];
  {
    const bf16* qp = Q + ((size_t)bh * S_LEN + q0 + rr) * HD + kb;
#pragma unroll
    for (int c = 0; c < 4; c++) {
      qf[c].q[0] = *(const uint4*)(qp + c * 32);
      qf[c].q[1] = *(const uint4*)(qp + c * 32 + 16);
    }
  }

  float mrun[8], lrun[8];
  v8f acc[8];
#pragma unroll
  for (int v = 0; v < 8; v++) { mrun[v] = -3.0e38f; lrun[v] = 0.f; }
#pragma unroll
  for (int cd = 0; cd < 8; cd++) {
    v8f z = {0.f, 0.f, 0.f, 0.f, 0.f, 0.f, 0.f, 0.f};
    acc[cd] = z;
  }

#if HAVE_TDM
  const unsigned ldsKoff = (unsigned)(uintptr_t)&Ks[0];
  const unsigned ldsVoff = (unsigned)(uintptr_t)&Vs[0];
#endif

  for (int kt = 0; kt < S_LEN; kt += 32) {
    __syncthreads();
#if HAVE_TDM
    if (wid == 0) {  // one wave drives the TDM for the whole workgroup
      tdm_load_k32xd128(Kk + ((size_t)bh * S_LEN + kt) * HD, ldsKoff);
      tdm_load_k32xd128(V  + ((size_t)bh * S_LEN + kt) * HD, ldsVoff);
      __builtin_amdgcn_s_wait_tensorcnt(0);
    }
#else
    {  // stage K,V 32x128 tiles
      int r = tid >> 3, cc = (tid & 7) * 16;
      const bf16* kp = Kk + ((size_t)bh * S_LEN + kt + r) * HD + cc;
      const bf16* vp = V  + ((size_t)bh * S_LEN + kt + r) * HD + cc;
      *(uint4*)&Ks[r * HD + cc]     = *(const uint4*)kp;
      *(uint4*)&Ks[r * HD + cc + 8] = *(const uint4*)(kp + 8);
      *(uint4*)&Vs[r * HD + cc]     = *(const uint4*)vp;
      *(uint4*)&Vs[r * HD + cc + 8] = *(const uint4*)(vp + 8);
    }
#endif
    __syncthreads();

    // scores: S = Q @ K^T  (two 16-key tiles)
    v8f sc[2];
#pragma unroll
    for (int j = 0; j < 2; j++) {
      v8f s = {0.f, 0.f, 0.f, 0.f, 0.f, 0.f, 0.f, 0.f};
#pragma unroll
      for (int c = 0; c < 4; c++) {
        Frag16 b;  // B[d][key]: contiguous along d in Ks row
        const bf16* p = &Ks[(j * 16 + nn) * HD + c * 32 + kb];
        b.q[0] = *(const uint4*)p;
        b.q[1] = *(const uint4*)(p + 16);
        s = WMMA_BF16(qf[c].v, b.v, s);
      }
      sc[j] = s;
    }

    // online softmax (rows live across the 16-lane half-wave)
#pragma unroll
    for (int v = 0; v < 8; v++) {
      float x0 = sc[0][v] * scale, x1 = sc[1][v] * scale;
      float t = half16_max(fmaxf(x0, x1));
      float mnew = fmaxf(mrun[v], t);
      float corr = __expf(mrun[v] - mnew);
      float p0 = __expf(x0 - mnew), p1 = __expf(x1 - mnew);
      lrun[v] = lrun[v] * corr + half16_sum(p0 + p1);  // denom: unmasked
      mrun[v] = mnew;
#pragma unroll
      for (int cd = 0; cd < 8; cd++) acc[cd][v] *= corr;
      // numerator weights get the mask (attn = softmax(scores) * mask)
      int qrow = q0 + rr8 + v;
      int kc   = kt + (lane & 15);
      p0 *= mask[(size_t)qrow * S_LEN + kc];
      p1 *= mask[(size_t)qrow * S_LEN + kc + 16];
      Ps[wid][(rr8 + v) * 32 + (lane & 15)]      = (bf16)p0;
      Ps[wid][(rr8 + v) * 32 + 16 + (lane & 15)] = (bf16)p1;
    }
    __syncthreads();

    // P (C layout) -> A fragment via LDS bounce
    Frag16 pa;
    {
      const bf16* pp = &Ps[wid][rr * 32];
#pragma unroll
      for (int v = 0; v < 8; v++) {
        int kk = kb16 + 2 * v;
        pa.u[v] = *(const unsigned*)(pp + kk);
      }
    }
    // O += P @ V
#pragma unroll
    for (int cd = 0; cd < 8; cd++) {
      Frag16 b;
      const bf16* p = &Vs[kb16 * HD + cd * 16 + nn];
#pragma unroll
      for (int e = 0; e < 16; e++) b.e[e] = p[e * HD];
      acc[cd] = WMMA_BF16(pa.v, b.v, acc[cd]);
    }
  }

  // normalize and scatter back to [B][S][HID] bf16
  const int b = bh >> 4, h = bh & 15;
#pragma unroll
  for (int v = 0; v < 8; v++) {
    float inv = 1.0f / lrun[v];
    int s = q0 + rr8 + v;
    size_t rowoff = ((size_t)b * S_LEN + s) * HID + h * HD;
#pragma unroll
    for (int cd = 0; cd < 8; cd++)
      O[rowoff + cd * 16 + (lane & 15)] = (bf16)(acc[cd][v] * inv);
  }
}

// ---------------------------------------------------------------------------
extern "C" void kernel_launch(void* const* d_in, const int* in_sizes, int n_in,
                              void* d_out, int out_size, void* d_ws,
                              size_t ws_size, hipStream_t stream) {
  (void)in_sizes; (void)n_in; (void)out_size; (void)ws_size;
  const float* x    = (const float*)d_in[0];
  const float* mask = (const float*)d_in[1];
  const float* Wq   = (const float*)d_in[2];
  const float* bq   = (const float*)d_in[3];
  const float* Wk   = (const float*)d_in[4];
  const float* bk   = (const float*)d_in[5];
  const float* Wv   = (const float*)d_in[6];
  const float* bv   = (const float*)d_in[7];
  const float* Wo   = (const float*)d_in[8];
  const float* bo   = (const float*)d_in[9];
  float* out = (float*)d_out;

  const size_t E  = (size_t)MROWS * HID;   // 16.7M
  const size_t WN = (size_t)HID * HID;     // 4.2M

  char* p = (char*)d_ws;
  bf16* xb  = (bf16*)p; p += E * 2;
  bf16* wqb = (bf16*)p; p += WN * 2;
  bf16* wkb = (bf16*)p; p += WN * 2;
  bf16* wvb = (bf16*)p; p += WN * 2;
  bf16* wob = (bf16*)p; p += WN * 2;
  bf16* qb  = (bf16*)p; p += E * 2;
  bf16* kb  = (bf16*)p; p += E * 2;
  bf16* vb  = (bf16*)p; p += E * 2;
  bf16* aob = (bf16*)p; p += E * 2;

  cast_f32_bf16<<<dim3((unsigned)((E + 255) / 256)), 256, 0, stream>>>(x, xb, E);
  cast_f32_bf16<<<dim3((unsigned)((WN + 255) / 256)), 256, 0, stream>>>(Wq, wqb, WN);
  cast_f32_bf16<<<dim3((unsigned)((WN + 255) / 256)), 256, 0, stream>>>(Wk, wkb, WN);
  cast_f32_bf16<<<dim3((unsigned)((WN + 255) / 256)), 256, 0, stream>>>(Wv, wvb, WN);
  cast_f32_bf16<<<dim3((unsigned)((WN + 255) / 256)), 256, 0, stream>>>(Wo, wob, WN);

  dim3 ggrid(MROWS / 64, HID / 256);
  proj_gemm<<<ggrid, 256, 0, stream>>>(xb, wqb, bq, qb, nullptr, 2);  // Q + RoPE
  proj_gemm<<<ggrid, 256, 0, stream>>>(xb, wkb, bk, kb, nullptr, 2);  // K + RoPE
  proj_gemm<<<ggrid, 256, 0, stream>>>(xb, wvb, bv, vb, nullptr, 1);  // V

  attn_kernel<<<dim3(BATCH * HEADS * (S_LEN / 128)), 256, 0, stream>>>(
      qb, kb, vb, mask, aob);

  proj_gemm<<<ggrid, 256, 0, stream>>>(aob, wob, bo, nullptr, out, 0);
}
